// MultiHeadAttention_72765335929011
// MI455X (gfx1250) — matrix operand order repaired and checked by reading, not compiled, not run on hardware
//
#include <hip/hip_runtime.h>
#include <math.h>

// Problem constants (match reference)
#define DM   256   // d_model
#define DKV  32    // per-head dim
#define NH   8     // heads
#define BB   16    // batch
#define LQn  128   // Lq
#define LKn  128   // Lk

typedef __attribute__((ext_vector_type(16))) __bf16 v16bf;
typedef __attribute__((ext_vector_type(8)))  __bf16 v8bf;
typedef __attribute__((ext_vector_type(4)))  __bf16 v4bf;
typedef __attribute__((ext_vector_type(8)))  float  v8f;

// ---- LDS layout (bytes). Total 158720 B < 160 KB -> 2 WGs / WGP. ----
#define OFF_A     0                      // bf16 sA[128][256] = 65536 B (h_arc tile, row-major k x i)
                                         //   later aliased as f32 sU[16][256] = 16384 B
#define OFF_AT    65536                  // bf16 sAT[256][TS] = 69632 B (i-major transposed copy)
#define TS        136                    // padded k-stride (bank-conflict avoidance)
#define OFF_QK    (OFF_AT + 69632)       // bf16 sQK[16][256] = 8192 B (head-major, heads 8..15 zero)
#define OFF_ATTN  (OFF_QK + 8192)        // bf16 sAttn[16][128] = 4096 B (heads 8..15 zero)
#define OFF_S     (OFF_ATTN + 4096)      // f32 sS[16][128] = 8192 B (scores, head-major)
#define OFF_H     (OFF_S + 8192)         // f32 sH[256]
#define OFF_Q     (OFF_H + 1024)         // f32 sQ[256]
#define OFF_C     (OFF_Q + 1024)         // f32 sC[256]
#define SMEM_BYTES (OFF_C + 1024)

// A-operand (16x32 bf16): lane holds row m = rowBase+ (lane&15).
// elems 0..7  = K[kBase + (lane>>4)*8 .. +7]
// elems 8..15 = K[kBase + 16 + (lane>>4)*8 .. +7]          (ISA 7.12.2)
static __device__ inline v16bf load_frag_a(const __bf16* src, int rowStride,
                                           int row, int kBase, int lhalf) {
  const __bf16* p = src + row * rowStride + kBase + lhalf * 8;
  union { v16bf v; v8bf h[2]; } u;
  u.h[0] = *(const v8bf*)(p);
  u.h[1] = *(const v8bf*)(p + 16);
  return u.v;
}

// B-operand (32x16 bf16) from an n-major [n][k] array:
// lane holds column n = nBase + (lane&15); elems 0..7 = K[kBase + (lane>>4)*8 .. +7], elems 8..15 = K[kBase + 16 + (lane>>4)*8 .. +7]
static __device__ inline v16bf load_frag_b(const __bf16* srcNmajor, int rowStride,
                                           int n, int kBase, int lhalf) {
  const __bf16* p = srcNmajor + n * rowStride + kBase + lhalf * 8;
  union { v16bf v; v8bf h[2]; } u;
  u.h[0] = *(const v8bf*)(p);
  u.h[1] = *(const v8bf*)(p + 16);
  return u.v;
}

__global__ __launch_bounds__(256) void mha_fused_kernel(
    const float* __restrict__ h,     const float* __restrict__ h_arc,
    const int*   __restrict__ mask,
    const float* __restrict__ Wq,    const float* __restrict__ bq,
    const float* __restrict__ Wk,    const float* __restrict__ bk,
    const float* __restrict__ Wv,    const float* __restrict__ bv,
    const float* __restrict__ Wo,    const float* __restrict__ bo,
    float* __restrict__ out)
{
  extern __shared__ __align__(16) unsigned char smem[];
  __bf16* sA    = (__bf16*)(smem + OFF_A);
  float*  sU    = (float*) (smem + OFF_A);    // alias: valid after scores stage
  __bf16* sAT   = (__bf16*)(smem + OFF_AT);
  __bf16* sQK   = (__bf16*)(smem + OFF_QK);
  __bf16* sAttn = (__bf16*)(smem + OFF_ATTN);
  float*  sS    = (float*) (smem + OFF_S);
  float*  sH    = (float*) (smem + OFF_H);
  float*  sQ    = (float*) (smem + OFF_Q);
  float*  sC    = (float*) (smem + OFF_C);

  const int tid   = threadIdx.x;
  const int lane  = tid & 31;
  const int wave  = tid >> 5;          // 0..7
  const int lmod  = lane & 15;
  const int lhalf = lane >> 4;
  const int bqi   = blockIdx.x;        // b*LQ + q

  // ---- Stage 0: h row to LDS; zero padded head rows of sQK / sAttn ----
  sH[tid] = h[(size_t)bqi * DM + tid];
#pragma unroll
  for (int j = 0; j < 8; ++j) sQK[NH * DM + tid * 8 + j] = (__bf16)0.0f;
#pragma unroll
  for (int j = 0; j < 4; ++j) sAttn[NH * LKn + tid * 4 + j] = (__bf16)0.0f;

  // ---- Stage 1: stream h_arc[b,q] (128x256 f32 = 128 KB), cvt -> bf16 in sA ----
  {
    const float4* src = (const float4*)(h_arc + (size_t)bqi * (LKn * DM));
#pragma unroll 4
    for (int it = 0; it < 32; ++it) {
      const int f  = it * 256 + tid;          // float4 index, coalesced
      const float4 v = src[f];
      const int k  = f >> 6;
      const int i0 = (f & 63) * 4;
      v4bf t;
      t[0] = (__bf16)v.x; t[1] = (__bf16)v.y;
      t[2] = (__bf16)v.z; t[3] = (__bf16)v.w;
      *(v4bf*)(sA + k * DM + i0) = t;
    }
  }
  __syncthreads();

  // ---- Stage 2a: transposed copy sA -> sAT (thread owns column i = tid) ----
  {
    const int i = tid;
    for (int kb = 0; kb < LKn; kb += 8) {
      v8bf t;
#pragma unroll
      for (int j = 0; j < 8; ++j) t[j] = sA[(kb + j) * DM + i];
      *(v8bf*)(sAT + i * TS + kb) = t;        // 272 B row stride: 16B aligned
    }
  }
  // ---- Stage 2b: q projection (o = tid): q[o] = h_row . Wq[o,:] + bq[o] ----
  {
    const int o = tid;
    float acc = bq[o];
    const float4* w4 = (const float4*)(Wq + (size_t)o * DM);   // L2-resident
    const float4* x4 = (const float4*)sH;
#pragma unroll 8
    for (int i2 = 0; i2 < DM / 4; ++i2) {
      const float4 w = w4[i2], x = x4[i2];
      acc += w.x * x.x + w.y * x.y + w.z * x.z + w.w * x.w;
    }
    sQ[o] = acc;
  }
  __syncthreads();

  // ---- Stage 3: qk[h,i] = sum_d q[h,d]*Wk[h*32+d, i]; fold 1/sqrt(256) ----
  // (bk contributes a per-head constant to scores; softmax is shift-invariant, so it cancels)
  {
    const int i = tid;
#pragma unroll
    for (int hh = 0; hh < NH; ++hh) {
      float acc = 0.0f;
      const float* wk = Wk + (size_t)(hh * DKV) * DM + i;      // coalesced across threads
#pragma unroll 8
      for (int d = 0; d < DKV; ++d) acc += sQ[hh * DKV + d] * wk[(size_t)d * DM];
      sQK[hh * DM + i] = (__bf16)(acc * 0.0625f);
    }
  }
  __syncthreads();

  // ---- Stage 4: scores = h_arc(128x256) @ qk^T(256x16) via bf16 WMMA ----
  // wave w owns M-tile of rows k = 16w..16w+15; K loop = 8 steps of 32
  {
    const int rowBase = wave * 16;
    v8f c = {};
#pragma unroll
    for (int kk = 0; kk < 8; ++kk) {
      const v16bf a = load_frag_a(sA,  DM, rowBase + lmod, kk * 32, lhalf);
      const v16bf b = load_frag_b(sQK, DM, lmod,           kk * 32, lhalf);
      c = __builtin_amdgcn_wmma_f32_16x16x32_bf16(false, a, false, b,
                                                  (short)0, c, false, false);
    }
#pragma unroll
    for (int r = 0; r < 8; ++r)                      // D: lane->(M=r+lhalf*8, N=lmod)
      sS[lmod * LKn + rowBase + r + lhalf * 8] = c[r];
  }
  __syncthreads();

  // ---- Stage 5: masked softmax per head (wave w == head w), wave32 shuffles ----
  {
    const int hh = wave;
    const int* mrow = mask + (size_t)bqi * LKn;
    float x[4]; int mm[4];
    float mx = -1e30f;
#pragma unroll
    for (int j = 0; j < 4; ++j) {
      const int k = lane + 32 * j;
      mm[j] = mrow[k];
      const float s = sS[hh * LKn + k];
      x[j] = (mm[j] == 0) ? -1e30f : s;
      mx = fmaxf(mx, x[j]);
    }
#pragma unroll
    for (int off = 16; off >= 1; off >>= 1) mx = fmaxf(mx, __shfl_xor(mx, off, 32));
    float e[4], sum = 0.0f;
#pragma unroll
    for (int j = 0; j < 4; ++j) {
      e[j] = (mm[j] == 0) ? 0.0f : __expf(x[j] - mx);
      sum += e[j];
    }
#pragma unroll
    for (int off = 16; off >= 1; off >>= 1) sum += __shfl_xor(sum, off, 32);
    const float inv = (sum > 0.0f) ? 1.0f / sum : 0.0f;
#pragma unroll
    for (int j = 0; j < 4; ++j)
      sAttn[hh * LKn + lane + 32 * j] = (__bf16)(e[j] * inv);
  }
  __syncthreads();

  // ---- Stage 6: u = attn(16x128) @ h_arc(128x256) via bf16 WMMA; D -> sU (aliases sA) ----
  {
#pragma unroll
    for (int t2 = 0; t2 < 2; ++t2) {
      const int nt = wave * 2 + t2;                  // N-tile 0..15
      v8f c = {};
#pragma unroll
      for (int kk = 0; kk < 4; ++kk) {
        const v16bf a = load_frag_a(sAttn, LKn, lmod,             kk * 32, lhalf);
        const v16bf b = load_frag_b(sAT,   TS,  nt * 16 + lmod,   kk * 32, lhalf);
        c = __builtin_amdgcn_wmma_f32_16x16x32_bf16(false, a, false, b,
                                                    (short)0, c, false, false);
      }
#pragma unroll
      for (int r = 0; r < 8; ++r)
        sU[(r + lhalf * 8) * DM + nt * 16 + lmod] = c[r];
    }
  }
  __syncthreads();

  // ---- Stage 7: ctx[o] = u[h,:].Wv[o,:] + bv[o]   (softmax rows sum to 1) ----
  {
    const int o = tid, hh = o >> 5;
    float acc = bv[o];
    const float4* w4 = (const float4*)(Wv + (size_t)o * DM);
    const float4* u4 = (const float4*)(sU + hh * DM);
#pragma unroll 8
    for (int i2 = 0; i2 < DM / 4; ++i2) {
      const float4 w = w4[i2], x = u4[i2];
      acc += w.x * x.x + w.y * x.y + w.z * x.z + w.w * x.w;
    }
    sC[o] = acc;
  }
  __syncthreads();

  // ---- Stage 8: out[o] = ctx . Wo[o,:] + bo[o] ----
  {
    const int o = tid;
    float acc = bo[o];
    const float4* w4 = (const float4*)(Wo + (size_t)o * DM);
    const float4* c4 = (const float4*)sC;
#pragma unroll 8
    for (int i2 = 0; i2 < DM / 4; ++i2) {
      const float4 w = w4[i2], x = c4[i2];
      acc += w.x * x.x + w.y * x.y + w.z * x.z + w.w * x.w;
    }
    out[(size_t)bqi * DM + o] = acc;
  }
}

extern "C" void kernel_launch(void* const* d_in, const int* in_sizes, int n_in,
                              void* d_out, int out_size, void* d_ws, size_t ws_size,
                              hipStream_t stream) {
  (void)in_sizes; (void)n_in; (void)out_size; (void)d_ws; (void)ws_size;
  const float* h     = (const float*)d_in[0];
  const float* h_arc = (const float*)d_in[1];
  const int*   mask  = (const int*)  d_in[2];
  const float* Wq    = (const float*)d_in[3];
  const float* bq    = (const float*)d_in[4];
  const float* Wk    = (const float*)d_in[5];
  const float* bk    = (const float*)d_in[6];
  const float* Wv    = (const float*)d_in[7];
  const float* bv    = (const float*)d_in[8];
  const float* Wo    = (const float*)d_in[9];
  const float* bo    = (const float*)d_in[10];
  float* out = (float*)d_out;

  (void)hipFuncSetAttribute((const void*)mha_fused_kernel,
                            hipFuncAttributeMaxDynamicSharedMemorySize, SMEM_BYTES);
  mha_fused_kernel<<<dim3(BB * LQn), dim3(256), SMEM_BYTES, stream>>>(
      h, h_arc, mask, Wq, bq, Wk, bk, Wv, bv, Wo, bo, out);
}
